// MessagePassingWithHalting_89988154786023
// MI455X (gfx1250) — matrix operand order repaired and checked by reading, not compiled, not run on hardware
//
#include <hip/hip_runtime.h>

#define NODES 4096
#define BATCH 8
#define DIM   256
#define MROWS (BATCH * NODES)            /* 32768 */
#define TOTAL ((size_t)MROWS * DIM)      /* 8388608 */

typedef __attribute__((ext_vector_type(16))) _Float16 v16h;
typedef __attribute__((ext_vector_type(8)))  _Float16 v8h;
typedef __attribute__((ext_vector_type(8)))  float    v8f;
typedef __attribute__((ext_vector_type(4)))  int      v4i;

#if defined(__HIP_DEVICE_COMPILE__) && \
    __has_builtin(__builtin_amdgcn_global_load_async_to_lds_b128) && \
    __has_builtin(__builtin_amdgcn_s_wait_asynccnt)
#define USE_ASYNC_LDS 1
#else
#define USE_ASYNC_LDS 0
#endif

__device__ __forceinline__ float gelu_f(float x) {
  return 0.5f * x * (1.0f + erff(x * 0.70710678118654752f));
}

__device__ __forceinline__ float block_sum_256(float v, float* sm) {
  #pragma unroll
  for (int off = 16; off > 0; off >>= 1) v += __shfl_xor(v, off, 32);
  const int w = threadIdx.x >> 5;
  __syncthreads();                      // protect sm from previous use
  if ((threadIdx.x & 31) == 0) sm[w] = v;
  __syncthreads();
  float t = 0.f;
  #pragma unroll
  for (int i = 0; i < 8; ++i) t += sm[i];
  return t;
}

// 16-byte global -> LDS copy; async-to-LDS on gfx1250, plain copy otherwise.
__device__ __forceinline__ void copy16_g2l(const _Float16* g, _Float16* l) {
#if USE_ASYNC_LDS
  // param1: AS1 (global) v4i*, param2: AS3 (LDS) v4i*.
  // Generic->AS pointers via integer round-trip: global generic == full VA;
  // LDS generic low 32 bits == LDS byte offset (ISA 10.2 aperture mapping).
  v4i __attribute__((address_space(1)))* gp =
      (v4i __attribute__((address_space(1)))*)(uintptr_t)g;
  v4i __attribute__((address_space(3)))* lp =
      (v4i __attribute__((address_space(3)))*)(uint32_t)(uintptr_t)l;
  __builtin_amdgcn_global_load_async_to_lds_b128(gp, lp, 0, 0);
#else
  *(uint4*)l = *(const uint4*)g;
#endif
}

__device__ __forceinline__ void staging_wait() {
#if USE_ASYNC_LDS
  __builtin_amdgcn_s_wait_asynccnt(0);
#endif
  __syncthreads();
}

// ---------------------------------------------------------------------------
// Weight conversion: f32 row-major (K x N) -> f16 WMMA B-fragment order,
// stored bank-conflict-free: each 512-half tile (kc,nt) is split into two
// half-fragments (e<8, e>=8), each lane-major at 16B/lane:
//   tile_off = (kc*NT + nt)*512 ; idx_in_tile = (e>>3)*256 + lane*8 + (e&7)
// with contiguous B k order (16h+e, used by this code) k = kc*32 + e + 16*(lane>>4), n = nt*16 + (lane&15).
// colSplit=1 handles msg_w1 (512x256) -> K=256 x N=512 column-concat:
//   n<256 : src[k*256+n] ; n>=256 : src[(k+256)*256 + (n-256)]
// ---------------------------------------------------------------------------
__global__ void conv_frag_kernel(const float* __restrict__ src, _Float16* __restrict__ dst,
                                 int K, int Nn, int colSplit) {
  const size_t idx = (size_t)blockIdx.x * blockDim.x + threadIdx.x;
  if (idx >= (size_t)K * Nn) return;
  const int elo  = (int)(idx & 7);
  const int lane = (int)((idx >> 3) & 31);
  const int hsel = (int)((idx >> 8) & 1);
  const size_t tile = idx >> 9;
  const int e  = hsel * 8 + elo;
  const int NT = Nn >> 4;
  const int nt = (int)(tile % NT);
  const int kc = (int)(tile / NT);
  const int k  = kc * 32 + e + 16 * (lane >> 4);
  const int n  = nt * 16 + (lane & 15);
  float v;
  if (colSplit) {
    v = (n < 256) ? src[(size_t)k * 256 + n] : src[(size_t)(k + 256) * 256 + (n - 256)];
  } else {
    v = src[(size_t)k * Nn + n];
  }
  dst[idx] = (_Float16)v;
}

// ---------------------------------------------------------------------------
// WMMA GEMM: C[m,n] = epi( sum_k A[m,k]*W[k,n] + bias[n] ), A row-major f32
// (ld = DIM), W in fragment layout, staged block-wide into LDS via async
// global->LDS loads. Dual-A: for kc >= KCsplit read A2 with rebased k
// (handles K-concat, e.g. [h || agg] @ upd_w1).
// Block: 256 thr = 8 waves; wave tile = 16 rows x 64 cols (4 WMMA tiles).
// ---------------------------------------------------------------------------
template <int KC>
__global__ __launch_bounds__(256)
void gemm_wmma_kernel(const float* __restrict__ A, const float* __restrict__ A2,
                      const _Float16* __restrict__ Wfrag, const float* __restrict__ bias,
                      float* __restrict__ C, int Nn, int KCsplit, int doGelu) {
  __shared__ __attribute__((aligned(32))) _Float16 lbuf[KC * 2048];

  const int tid   = threadIdx.x;
  const int lane  = tid & 31;
  const int wave  = tid >> 5;
  const int half  = lane >> 4;
  const int l15   = lane & 15;
  const int mbase = blockIdx.x * 128 + wave * 16;
  const int ntbase = blockIdx.y * 4;
  const int NT = Nn >> 4;

  // Stage this block's B fragments (KC k-chunks x 4 n-tiles, 512 halves each).
  // LDS mirrors the global fragment order within each 2048-half kc-chunk.
  #pragma unroll
  for (int kc = 0; kc < KC; ++kc) {
    const size_t goff = (((size_t)(kc * NT + ntbase)) << 9) + (tid << 3);
    copy16_g2l(Wfrag + goff, lbuf + (kc << 11) + (tid << 3));
  }
  staging_wait();

  v8f acc[4];
  #pragma unroll
  for (int t = 0; t < 4; ++t) acc[t] = (v8f){};

  const int arow = mbase + l15;
  for (int kc = 0; kc < KC; ++kc) {
    const float* Ap = (kc < KCsplit) ? A : A2;
    const int kk = ((kc < KCsplit) ? kc : (kc - KCsplit)) * 32;
    // A fragment (contiguous, matching B): e -> k=kk+16h+e
    const float* base = Ap + (size_t)arow * DIM + kk + half * 16;
    const float4 f0 = *(const float4*)(base + 0);
    const float4 f1 = *(const float4*)(base + 4);
    const float4 f2 = *(const float4*)(base + 8);
    const float4 f3 = *(const float4*)(base + 12);
    v16h a;
    a[0] = (_Float16)f0.x; a[1] = (_Float16)f0.y; a[2]  = (_Float16)f0.z; a[3]  = (_Float16)f0.w;
    a[4] = (_Float16)f1.x; a[5] = (_Float16)f1.y; a[6]  = (_Float16)f1.z; a[7]  = (_Float16)f1.w;
    a[8] = (_Float16)f2.x; a[9] = (_Float16)f2.y; a[10] = (_Float16)f2.z; a[11] = (_Float16)f2.w;
    a[12] = (_Float16)f3.x; a[13] = (_Float16)f3.y; a[14] = (_Float16)f3.z; a[15] = (_Float16)f3.w;

    // Batch all 4 B-fragment loads (2 x v8h each, 16B/lane -> 2-way ideal
    // bank pattern) before the 4 WMMAs, so waits can be partial.
    v8h blo[4], bhi[4];
    #pragma unroll
    for (int t = 0; t < 4; ++t) {
      const _Float16* tb = lbuf + (kc << 11) + (t << 9);
      blo[t] = *(const v8h*)(tb + (lane << 3));
      bhi[t] = *(const v8h*)(tb + 256 + (lane << 3));
    }
    #pragma unroll
    for (int t = 0; t < 4; ++t) {
      v16h b;
      #pragma unroll
      for (int e = 0; e < 8; ++e) { b[e] = blo[t][e]; b[8 + e] = bhi[t][e]; }
      acc[t] = __builtin_amdgcn_wmma_f32_16x16x32_f16(
          false, a, false, b, (short)0, acc[t], false, false);
    }
  }
  // C/D layout: VGPR v -> row = mbase + v + 8*half, col = nt*16 + (lane&15)
  #pragma unroll
  for (int t = 0; t < 4; ++t) {
    const int col = (ntbase + t) * 16 + l15;
    const float bv = bias[col];
    #pragma unroll
    for (int v = 0; v < 8; ++v) {
      const int row = mbase + v + 8 * half;
      float x = acc[t][v] + bv;
      if (doGelu) x = gelu_f(x);
      C[(size_t)row * Nn + col] = x;
    }
  }
}

// G[b,n,d] = (sum_{o=0..3, n-o>=0} gelu(S[b,n,d] + Ng[b,n-o,d])) / cnt(n)
// SN rows are [S(256, incl. msg_b1 via bias) || Ng(256)]
__global__ __launch_bounds__(256)
void gelu_agg_kernel(const float* __restrict__ SN, float* __restrict__ G) {
  const int m = blockIdx.x;
  const int d = threadIdx.x;
  const int n = m & (NODES - 1);
  const float s = SN[(size_t)m * 512 + d];
  float acc = 0.f; int cnt = 0;
  #pragma unroll
  for (int o = 0; o < 4; ++o) {
    if (n - o >= 0) {
      const float ng = SN[(size_t)(m - o) * 512 + 256 + d];
      acc += gelu_f(s + ng);
      ++cnt;
    }
  }
  G[(size_t)m * DIM + d] = acc / (float)cnt;
}

// h_new = LN(h + Dlt); also per-batch column sums (for halting mean over nodes)
__global__ __launch_bounds__(256)
void ln_colsum_kernel(const float* __restrict__ h, const float* __restrict__ dlt,
                      const float* __restrict__ g, const float* __restrict__ b,
                      float* __restrict__ hn, float* __restrict__ colsum) {
  __shared__ float sm[8];
  const int d = threadIdx.x;
  const int row0 = blockIdx.x * 16;
  const int bat = row0 / NODES;
  const float gd = g[d], bd = b[d];
  float accd = 0.f;
  for (int i = 0; i < 16; ++i) {
    const size_t idx = (size_t)(row0 + i) * DIM + d;
    const float x  = h[idx] + dlt[idx];
    const float mu = block_sum_256(x, sm) * (1.f / 256.f);
    const float xc = x - mu;
    const float var = block_sum_256(xc * xc, sm) * (1.f / 256.f);
    const float y = xc * rsqrtf(var + 1e-5f) * gd + bd;
    hn[idx] = y;
    accd += y;
  }
  atomicAdd(&colsum[bat * DIM + d], accd);
}

__global__ void zero_colsum_kernel(float* __restrict__ colsum) {
  colsum[blockIdx.x * 256 + threadIdx.x] = 0.f;
}

// scal layout: [0..7]=remaining, [8..15]=sw, [16]=kl, [17]=active, [18]=commit
__global__ __launch_bounds__(256)
void halt_kernel(const float* __restrict__ colsum, const float* __restrict__ halt_w,
                 const float* __restrict__ halt_b, float* __restrict__ scal, int step) {
  __shared__ float sm[8];
  const int d = threadIdx.x;
  float halts[BATCH];
  const float hw = halt_w[d];
  for (int bb = 0; bb < BATCH; ++bb) {
    const float v = colsum[bb * DIM + d] * (1.f / (float)NODES) * hw;
    const float tot = block_sum_256(v, sm);
    float hl = 1.f / (1.f + expf(-(tot + halt_b[0])));
    if (step == 7) hl = 1.f;
    halts[bb] = hl;
  }
  if (d == 0) {
    const float active = scal[17];
    scal[18] = active;                  // commit this round's update?
    if (active > 0.5f) {
      float q = 0.f;
      for (int bb = 0; bb < BATCH; ++bb) q += halts[bb];
      q = q * (1.f / BATCH) + 1e-8f;
      const float p = 0.2f * powf(0.8f, (float)step);
      scal[16] += p * (logf(p) - logf(q));
      float mx = 0.f;
      for (int bb = 0; bb < BATCH; ++bb) {
        const float rem = scal[bb];
        scal[8 + bb] = rem * halts[bb];         // sw (uses old remaining)
        const float rn = rem * (1.f - halts[bb]);
        scal[bb] = rn;
        mx = fmaxf(mx, rn);
      }
      if (mx < 0.01f) scal[17] = 0.f;
    } else {
      for (int bb = 0; bb < BATCH; ++bb) scal[8 + bb] = 0.f;
    }
  }
}

__global__ void commit_kernel(float* __restrict__ h, const float* __restrict__ hn,
                              float* __restrict__ out, const float* __restrict__ scal) {
  const size_t i = (size_t)blockIdx.x * blockDim.x + threadIdx.x;
  if (i >= TOTAL) return;
  if (scal[18] > 0.5f) {
    const int bat = (int)(i / ((size_t)NODES * DIM));
    const float v = hn[i];
    h[i] = v;
    out[i] += scal[8 + bat] * v;
  }
}

__global__ void init_kernel(const float* __restrict__ summaries, float* __restrict__ h,
                            float* __restrict__ out, float* __restrict__ scal,
                            const float* __restrict__ msg_b1, float* __restrict__ bias512) {
  const size_t i = (size_t)blockIdx.x * blockDim.x + threadIdx.x;
  if (i < TOTAL) { h[i] = summaries[i]; out[i] = 0.f; }
  if (i < 512) bias512[i] = (i < 256) ? msg_b1[i] : 0.f;
  if (i == 0) {
    for (int bb = 0; bb < BATCH; ++bb) { scal[bb] = 1.f; scal[8 + bb] = 0.f; }
    scal[16] = 0.f; scal[17] = 1.f; scal[18] = 0.f;
  }
}

__global__ void finalize_kernel(const float* __restrict__ scal, float* __restrict__ out) {
  out[TOTAL] = scal[16] * (1.f / 8.f);
}

extern "C" void kernel_launch(void* const* d_in, const int* in_sizes, int n_in,
                              void* d_out, int out_size, void* d_ws, size_t ws_size,
                              hipStream_t stream) {
  (void)in_sizes; (void)n_in; (void)out_size; (void)ws_size;
  const float* summaries = (const float*)d_in[0];
  const float* msg_w1 = (const float*)d_in[1];
  const float* msg_b1 = (const float*)d_in[2];
  const float* msg_w2 = (const float*)d_in[3];
  const float* msg_b2 = (const float*)d_in[4];
  const float* upd_w1 = (const float*)d_in[5];
  const float* upd_b1 = (const float*)d_in[6];
  const float* upd_w2 = (const float*)d_in[7];
  const float* upd_b2 = (const float*)d_in[8];
  const float* halt_w = (const float*)d_in[9];
  const float* halt_b = (const float*)d_in[10];
  const float* ln_g   = (const float*)d_in[11];
  const float* ln_b   = (const float*)d_in[12];
  float* out = (float*)d_out;

  // Workspace layout (floats)
  float* ws     = (float*)d_ws;
  float* h      = ws;                      // 8388608
  float* hn     = h  + TOTAL;              // 8388608
  float* SN     = hn + TOTAL;              // 16777216  (S||Ng, later U)
  float* G      = SN + 2 * TOTAL;          // 8388608   (G, later Dlt)
  float* agg    = G  + TOTAL;              // 8388608
  float* colsum = agg + TOTAL;             // 2048
  float* scal   = colsum + BATCH * DIM;    // 64
  float* bias512 = scal + 64;              // 512
  _Float16* Wm1 = (_Float16*)(bias512 + 512);  // 131072 halves (K=256,N=512)
  _Float16* Wm2 = Wm1 + 131072;                // 65536  (K=256,N=256)
  _Float16* Wu1 = Wm2 + 65536;                 // 131072 (K=512,N=256)
  _Float16* Wu2 = Wu1 + 131072;                // 65536  (K=256,N=256)
  float* U   = SN;   // reuse after gelu_agg consumes SN
  float* Dlt = G;    // reuse after GEMM_B consumes G

  const int nblk = (int)(TOTAL / 256);     // 32768

  init_kernel<<<nblk, 256, 0, stream>>>(summaries, h, out, scal, msg_b1, bias512);
  conv_frag_kernel<<<131072 / 256, 256, 0, stream>>>(msg_w1, Wm1, 256, 512, 1);
  conv_frag_kernel<<<65536 / 256, 256, 0, stream>>>(msg_w2, Wm2, 256, 256, 0);
  conv_frag_kernel<<<131072 / 256, 256, 0, stream>>>(upd_w1, Wu1, 512, 256, 0);
  conv_frag_kernel<<<65536 / 256, 256, 0, stream>>>(upd_w2, Wu2, 256, 256, 0);

  const dim3 gA(MROWS / 128, 512 / 64);    // N=512
  const dim3 g2(MROWS / 128, 256 / 64);    // N=256

  for (int r = 0; r < 8; ++r) {
    // S||Ng = h @ [msg_w1_top || msg_w1_bot] (+ msg_b1 on S columns)
    gemm_wmma_kernel<8><<<gA, 256, 0, stream>>>(h, h, Wm1, bias512, SN, 512, 8, 0);
    // windowed masked-mean of gelu(S + Ng), pre-W2
    gelu_agg_kernel<<<MROWS, 256, 0, stream>>>(SN, G);
    // agg = G @ msg_w2 + msg_b2
    gemm_wmma_kernel<8><<<g2, 256, 0, stream>>>(G, G, Wm2, msg_b2, agg, 256, 8, 0);
    // U = gelu([h || agg] @ upd_w1 + upd_b1)   (dual-A, K=512)
    gemm_wmma_kernel<16><<<g2, 256, 0, stream>>>(h, agg, Wu1, upd_b1, U, 256, 8, 1);
    // Dlt = U @ upd_w2 + upd_b2
    gemm_wmma_kernel<8><<<g2, 256, 0, stream>>>(U, U, Wu2, upd_b2, Dlt, 256, 8, 0);
    // h_new = LN(h + Dlt); per-batch column sums
    zero_colsum_kernel<<<BATCH, 256, 0, stream>>>(colsum);
    ln_colsum_kernel<<<MROWS / 16, 256, 0, stream>>>(h, Dlt, ln_g, ln_b, hn, colsum);
    // halting scalars + gated state commit
    halt_kernel<<<1, 256, 0, stream>>>(colsum, halt_w, halt_b, scal, r);
    commit_kernel<<<nblk, 256, 0, stream>>>(h, hn, out, scal);
  }
  finalize_kernel<<<1, 1, 0, stream>>>(scal, out);
}
